// Attention_79121887527485
// MI455X (gfx1250) — matrix operand order repaired and checked by reading, not compiled, not run on hardware
//
#include <hip/hip_runtime.h>
#include <hip/hip_bf16.h>
#include <math.h>

// Problem constants (from reference): B=2, S=2048, D=2048, H=16, DH=128
#define BB   2
#define SS   2048
#define DD   2048
#define HH   16
#define DHH  128

typedef __attribute__((ext_vector_type(16))) __bf16       v16bf;
typedef __attribute__((ext_vector_type(8)))  float        v8f;
typedef __attribute__((ext_vector_type(4)))  unsigned int v4u;

union Frag { v16bf h; v4u u[2]; };

__device__ __forceinline__ __bf16 f2bf(float f) {
    union { float f; unsigned u; } a; a.f = f;
    unsigned r = a.u + 0x7FFFu + ((a.u >> 16) & 1u);   // round-to-nearest-even
    unsigned short hs = (unsigned short)(r >> 16);
    union { unsigned short s; __bf16 b; } o; o.s = hs; return o.b;
}

// ---- CDNA5 async copy: global -> LDS, 16B per lane, tracked by ASYNCcnt ---
__device__ __forceinline__ void async_cp16(unsigned lds_off, const __bf16* gptr) {
    asm volatile("global_load_async_to_lds_b128 %0, %1, off"
                 :: "v"(lds_off), "v"(gptr) : "memory");
}
__device__ __forceinline__ void wait_async3() {
    asm volatile("s_wait_asynccnt 0x3" ::: "memory");
}
__device__ __forceinline__ void wait_async0() {
    asm volatile("s_wait_asynccnt 0x0" ::: "memory");
}

// ---- WMMA fragment loaders (ISA 7.12.2 layouts, wave32) -------------------
// 16-bit A 16x32: lane = M (0..15, repeated); lanes<16 hold K {0..7,16..23},
// lanes>=16 hold K {8..15,24..31}. Two contiguous 16B chunks per lane.
__device__ __forceinline__ v16bf load_a16(const __bf16* base, int ld, int row0,
                                          int k0, int lane) {
    int m  = lane & 15;
    int hl = lane >> 4;
    const __bf16* p = base + (size_t)(row0 + m) * ld + k0 + hl * 8;
    Frag f;
    f.u[0] = *reinterpret_cast<const v4u*>(p);        // K = k0+8*hl .. +7
    f.u[1] = *reinterpret_cast<const v4u*>(p + 16);   // K = k0+16+8*hl .. +7
    return f.h;
}

// 16-bit B 32x16 where wmma-B[k][n] = W[col0+n][k0+k] (W row-major [N x K]).
// lane = N (0..15, repeated); lanes<16 hold K {0..7,16..23}, lanes>=16 hold K {8..15,24..31}.
__device__ __forceinline__ v16bf load_bt16(const __bf16* base, int ld, int col0,
                                           int k0, int lane) {
    int n  = lane & 15;
    int hl = lane >> 4;
    const __bf16* p = base + (size_t)(col0 + n) * ld + k0 + hl * 8;
    Frag f;
    f.u[0] = *reinterpret_cast<const v4u*>(p);
    f.u[1] = *reinterpret_cast<const v4u*>(p + 16);
    return f.h;
}

// Same patterns but reading a [rows x 32] bf16 tile that lives in LDS.
__device__ __forceinline__ v16bf lds_a16(const __bf16* tile, int row0, int lane) {
    int m  = lane & 15;
    int hl = lane >> 4;
    const v4u* p = reinterpret_cast<const v4u*>(tile + (row0 + m) * 32 + hl * 8);
    Frag f;
    f.u[0] = p[0];
    f.u[1] = p[2];    // +16 elements
    return f.h;
}
__device__ __forceinline__ v16bf lds_bt16(const __bf16* tile, int col0, int lane) {
    int n  = lane & 15;
    int hl = lane >> 4;
    const v4u* p = reinterpret_cast<const v4u*>(tile + (col0 + n) * 32 + hl * 8);
    Frag f;
    f.u[0] = p[0];
    f.u[1] = p[2];    // +16 elements
    return f.h;
}

// ---- elementwise fp32 -> bf16 ---------------------------------------------
__global__ void cvt_bf16_kernel(const float* __restrict__ in,
                                __bf16* __restrict__ out, long n) {
    long i = (long)blockIdx.x * blockDim.x + threadIdx.x;
    long st = (long)gridDim.x * blockDim.x;
    for (; i < n; i += st) out[i] = f2bf(in[i]);
}

// ---- GEMM: C[M,N] = A[M,K] * Bm[N,K]^T, bf16 in, f32 accumulate -----------
// Block tile 128x64, K-step 32, double-buffered LDS fed by async global->LDS.
// mode 0: f32 out row-major [M,N]
// mode 1: bf16 out head-major [B,H,S,DH] (Q: scale=1/sqrt(DH); K: scale=1)
// mode 2: bf16 out transposed head-major [B,H,DH,S] (V)
__global__ __launch_bounds__(256)
void gemm_bf16_kernel(const __bf16* __restrict__ A, const __bf16* __restrict__ Bm,
                      void* __restrict__ Out, int M, int N, int Kdim,
                      int mode, float scale) {
    // lds[buf]: A-tile [128][32] at 0, B-tile [64][32] at 128*32
    __shared__ __bf16 lds[2][(128 + 64) * 32];          // 24 KB

    const int t    = threadIdx.x;
    const int lane = t & 31;
    const int wv   = t >> 5;                            // 8 waves, 16 rows each
    const int row0 = blockIdx.y * 128;
    const int col0 = blockIdx.x * 64;

    const unsigned lbase   = (unsigned)(unsigned long long)(const void*)&lds[0][0];
    const unsigned bufStride = (128 + 64) * 32 * 2;     // bytes
    const unsigned bOff      = 128 * 32 * 2;            // B-tile byte offset

    // per-thread async copy slots: 2 chunks of A (32B), 1 chunk of B (16B)
    const int ca0 = t * 2, ca1 = t * 2 + 1, cb = t;
    const int ra0 = ca0 >> 2, ea0 = (ca0 & 3) * 8;
    const int ra1 = ca1 >> 2, ea1 = (ca1 & 3) * 8;
    const int rb  = cb  >> 2, eb  = (cb  & 3) * 8;

    auto issue = [&](int k0, int buf) {
        unsigned la = lbase + (unsigned)buf * bufStride;
        unsigned lb = la + bOff;
        async_cp16(la + ca0 * 16, A  + (size_t)(row0 + ra0) * Kdim + k0 + ea0);
        async_cp16(la + ca1 * 16, A  + (size_t)(row0 + ra1) * Kdim + k0 + ea1);
        async_cp16(lb + cb  * 16, Bm + (size_t)(col0 + rb ) * Kdim + k0 + eb );
    };

    v8f acc[4];
    #pragma unroll
    for (int f = 0; f < 4; ++f) acc[f] = (v8f){};

    const int nK = Kdim / 32;
    issue(0, 0);
    for (int kt = 0; kt < nK; ++kt) {
        const int buf = kt & 1;
        if (kt + 1 < nK) { issue((kt + 1) * 32, buf ^ 1); wait_async3(); }
        else             { wait_async0(); }
        __syncthreads();                                 // stage kt visible

        const __bf16* Ab = &lds[buf][0];
        const __bf16* Bb = &lds[buf][128 * 32];
        v16bf a = lds_a16(Ab, wv * 16, lane);
        #pragma unroll
        for (int f = 0; f < 4; ++f) {
            v16bf b = lds_bt16(Bb, 16 * f, lane);
            acc[f] = __builtin_amdgcn_wmma_f32_16x16x32_bf16(
                false, a, false, b, (short)0, acc[f], false, false);
        }
        __syncthreads();                                 // reads done before reuse
    }

    const int n  = lane & 15;
    const int hl = lane >> 4;
    const int wrow0 = row0 + wv * 16;
    if (mode == 0) {
        float* Of = (float*)Out;
        #pragma unroll
        for (int f = 0; f < 4; ++f)
            #pragma unroll
            for (int r = 0; r < 8; ++r) {
                int m = r + 8 * hl;
                Of[(size_t)(wrow0 + m) * N + col0 + 16 * f + n] = acc[f][r];
            }
    } else {
        __bf16* Ob = (__bf16*)Out;
        #pragma unroll
        for (int f = 0; f < 4; ++f)
            #pragma unroll
            for (int r = 0; r < 8; ++r) {
                int m    = r + 8 * hl;
                int row  = wrow0 + m;         // = b*S + s
                int bidx = row / SS, s = row % SS;
                int ncol = col0 + 16 * f + n; // = h*DH + d
                int h = ncol / DHH, d = ncol % DHH;
                float v = acc[f][r] * scale;
                size_t idx = (mode == 1)
                    ? (((size_t)(bidx * HH + h) * SS + s) * DHH + d)
                    : (((size_t)(bidx * HH + h) * DHH + d) * SS + s);
                Ob[idx] = f2bf(v);
            }
    }
}

// ---- Flash attention with ALiBi + causal mask -----------------------------
// Q,K: bf16 [B,H,S,DH] (Q pre-scaled by 1/sqrt(DH)); Vt: bf16 [B,H,DH,S].
// Z out: bf16 [B,S,H*DH]. One wave owns a 16-query tile; 32 keys per step.
__global__ __launch_bounds__(128)
void attn_kernel(const __bf16* __restrict__ Q, const __bf16* __restrict__ K,
                 const __bf16* __restrict__ Vt, __bf16* __restrict__ Z) {
    __shared__ __bf16 pbuf[4][16 * 32];           // per-wave P staging (1 KB)

    const int lane = threadIdx.x & 31;
    const int wv   = threadIdx.x >> 5;
    const int bh   = blockIdx.y;                  // b*H + h
    const int b    = bh / HH, h = bh % HH;
    const int qb   = blockIdx.x * 64 + wv * 16;   // query tile base
    const float slope = exp2f(-0.5f * (float)(h + 1));  // ALiBi, H=16 pow2

    const __bf16* Qp = Q  + (size_t)bh * SS * DHH;
    const __bf16* Kp = K  + (size_t)bh * SS * DHH;
    const __bf16* Vp = Vt + (size_t)bh * DHH * SS;

    const int n  = lane & 15;
    const int hl = lane >> 4;
    __bf16* pb = pbuf[wv];

    v16bf aq[4];
    #pragma unroll
    for (int c = 0; c < 4; ++c) aq[c] = load_a16(Qp, DHH, qb, 32 * c, lane);

    v8f o[8];
    #pragma unroll
    for (int f = 0; f < 8; ++f) o[f] = (v8f){};
    float mrow[8], lrow[8];
    #pragma unroll
    for (int r = 0; r < 8; ++r) { mrow[r] = -3.0e38f; lrow[r] = 0.0f; }

    for (int j0 = 0; j0 < qb + 16; j0 += 32) {
        // ---- S = Q K^T over DH=128 (4 x k32), 32 keys as two C fragments
        v8f s0 = (v8f){}, s1 = (v8f){};
        #pragma unroll
        for (int c = 0; c < 4; ++c) {
            v16bf bk0 = load_bt16(Kp, DHH, j0,      32 * c, lane);
            v16bf bk1 = load_bt16(Kp, DHH, j0 + 16, 32 * c, lane);
            s0 = __builtin_amdgcn_wmma_f32_16x16x32_bf16(
                false, aq[c], false, bk0, (short)0, s0, false, false);
            s1 = __builtin_amdgcn_wmma_f32_16x16x32_bf16(
                false, aq[c], false, bk1, (short)0, s1, false, false);
        }

        // ---- clamp + ALiBi + causal mask; online softmax stats per row
        float alpha[8];
        #pragma unroll
        for (int r = 0; r < 8; ++r) {
            int m = r + 8 * hl;
            int i = qb + m;
            int ja = j0 + n, jb = j0 + 16 + n;
            float x0 = fminf(fmaxf(s0[r], -1024.f), 1024.f) + slope * (float)ja;
            float x1 = fminf(fmaxf(s1[r], -1024.f), 1024.f) + slope * (float)jb;
            if (ja > i) x0 = -3.0e38f;
            if (jb > i) x1 = -3.0e38f;
            float tv = fmaxf(x0, x1);                      // row max over 16 lanes
            tv = fmaxf(tv, __shfl_xor(tv, 1));
            tv = fmaxf(tv, __shfl_xor(tv, 2));
            tv = fmaxf(tv, __shfl_xor(tv, 4));
            tv = fmaxf(tv, __shfl_xor(tv, 8));
            float mnew = fmaxf(mrow[r], tv);
            alpha[r] = __expf(mrow[r] - mnew);
            float p0 = __expf(x0 - mnew);
            float p1 = __expf(x1 - mnew);
            float rs = p0 + p1;
            rs += __shfl_xor(rs, 1);
            rs += __shfl_xor(rs, 2);
            rs += __shfl_xor(rs, 4);
            rs += __shfl_xor(rs, 8);
            lrow[r] = lrow[r] * alpha[r] + rs;
            mrow[r] = mnew;
            // stage P (row-major 16x32 bf16) for C-layout -> A-layout relayout
            pb[m * 32 + n]      = f2bf(p0);
            pb[m * 32 + 16 + n] = f2bf(p1);
        }
        #pragma unroll
        for (int f = 0; f < 8; ++f)
            #pragma unroll
            for (int r = 0; r < 8; ++r) o[f][r] *= alpha[r];

        // LDS is in-order within a wave; wait for the P stores, then gather
        asm volatile("s_wait_dscnt 0x0" ::: "memory");
        Frag fp;
        {
            const v4u* ps = reinterpret_cast<const v4u*>(pb + (lane & 15) * 32 + hl * 8);
            fp.u[0] = ps[0];   // K = 8*hl .. +7
            fp.u[1] = ps[2];   // K = 16+8*hl .. +7
        }
        asm volatile("" ::: "memory");

        // ---- O += P * V  (V pre-transposed: contiguous key runs per dh row)
        #pragma unroll
        for (int f = 0; f < 8; ++f) {
            v16bf bv = load_bt16(Vp, SS, 16 * f, j0, lane);
            o[f] = __builtin_amdgcn_wmma_f32_16x16x32_bf16(
                false, fp.h, false, bv, (short)0, o[f], false, false);
        }
    }

    // ---- normalize and write Z as [B,S,H*DH] bf16
    #pragma unroll
    for (int f = 0; f < 8; ++f)
        #pragma unroll
        for (int r = 0; r < 8; ++r) {
            int m = r + 8 * hl;
            float val = o[f][r] / lrow[r];
            size_t row = (size_t)b * SS + (qb + m);
            int col = h * DHH + 16 * f + n;
            Z[row * (size_t)DD + col] = f2bf(val);
        }
}

// ---------------------------------------------------------------------------
extern "C" void kernel_launch(void* const* d_in, const int* in_sizes, int n_in,
                              void* d_out, int out_size, void* d_ws, size_t ws_size,
                              hipStream_t stream) {
    const float* X  = (const float*)d_in[0];  // [B,S,D]
    const float* Wq = (const float*)d_in[1];  // [H*DH, D]
    const float* Wk = (const float*)d_in[2];
    const float* Wv = (const float*)d_in[3];
    const float* Wo = (const float*)d_in[4];  // [D, H*DH]

    const size_t nX = (size_t)BB * SS * DD;   // 8,388,608
    const size_t nW = (size_t)DD * DD;        // 4,194,304

    char* ws = (char*)d_ws;
    size_t off = 0;
    __bf16* Xh  = (__bf16*)(ws + off); off += nX * 2;
    __bf16* Wqh = (__bf16*)(ws + off); off += nW * 2;
    __bf16* Wkh = (__bf16*)(ws + off); off += nW * 2;
    __bf16* Wvh = (__bf16*)(ws + off); off += nW * 2;
    __bf16* Woh = (__bf16*)(ws + off); off += nW * 2;
    __bf16* Qh  = (__bf16*)(ws + off); off += nX * 2;  // [B,H,S,DH]
    __bf16* Kh  = (__bf16*)(ws + off); off += nX * 2;  // [B,H,S,DH]
    __bf16* Vh  = (__bf16*)(ws + off); off += nX * 2;  // [B,H,DH,S]
    __bf16* Zh  = (__bf16*)(ws + off); off += nX * 2;  // [B,S,H*DH]

    // 1) fp32 -> bf16 conversions
    cvt_bf16_kernel<<<1024, 256, 0, stream>>>(X,  Xh,  (long)nX);
    cvt_bf16_kernel<<<512,  256, 0, stream>>>(Wq, Wqh, (long)nW);
    cvt_bf16_kernel<<<512,  256, 0, stream>>>(Wk, Wkh, (long)nW);
    cvt_bf16_kernel<<<512,  256, 0, stream>>>(Wv, Wvh, (long)nW);
    cvt_bf16_kernel<<<512,  256, 0, stream>>>(Wo, Woh, (long)nW);

    // 2) QKV projections (WMMA GEMM + async LDS staging, fused epilogues)
    dim3 gg(DD / 64, (BB * SS) / 128);        // (32, 32)
    const float qscale = 0.08838834764831845f;  // 1/sqrt(DH)
    gemm_bf16_kernel<<<gg, 256, 0, stream>>>(Xh, Wqh, Qh, BB * SS, DD, DD, 1, qscale);
    gemm_bf16_kernel<<<gg, 256, 0, stream>>>(Xh, Wkh, Kh, BB * SS, DD, DD, 1, 1.0f);
    gemm_bf16_kernel<<<gg, 256, 0, stream>>>(Xh, Wvh, Vh, BB * SS, DD, DD, 2, 1.0f);

    // 3) flash attention with ALiBi + causal
    attn_kernel<<<dim3(SS / 64, BB * HH), 128, 0, stream>>>(Qh, Kh, Vh, Zh);

    // 4) output projection -> fp32 d_out [B,S,D]
    gemm_bf16_kernel<<<gg, 256, 0, stream>>>(Zh, Woh, d_out, BB * SS, DD, DD, 0, 1.0f);
}
